// ParticleEnergyPredictor_New_81836306858288
// MI455X (gfx1250) — matrix operand order repaired and checked by reading, not compiled, not run on hardware
//
#include <hip/hip_runtime.h>
#include <hip/hip_bf16.h>

typedef __attribute__((ext_vector_type(16))) _Float16 v16h;
typedef __attribute__((ext_vector_type(8)))  _Float16 v8h;
typedef __attribute__((ext_vector_type(8)))  float    v8f;

#define WAVES 4
#define TPB   (WAVES * 32)
#define TPW   8                      // row-tiles (of 16) per wave
#define B_    2048
#define NB    250
#define ROWS  (B_ * NB)
#define NTILES (ROWS / 16)           // 32000
#define NBLOCKS (NTILES / (WAVES * TPW))   // 1000
#define EPSC  1e-8f
#define BNEPS 1e-5f
#define LOG2E 1.4426950408889634f

// ---------------- workspace layout (floats) ----------------
#define WS_BNP_SUM 0
#define WS_BNP_SQ  256
#define WS_BNE_SUM 512
#define WS_BNE_SQ  768
#define WS_BNP_A   1024
#define WS_BNP_C   1280
#define WS_BNE_A   1536
#define WS_BNE_C   1792
#define WS_S       2047
#define WS_PRIOR   2048
#define WS_GO      4096
#define WS_TOTAL   (4096 + B_ * 9)   // 22528 floats

// ---------------- dynamic LDS layout ----------------
// f32 section (float indices)
#define SF_PB0 0
#define SF_PB1 64
#define SF_EB0 128
#define SF_EB1 256
#define SF_PW2 384
#define SF_EW2 448
#define SF_SC  576          // [0]=p_b2 [1]=e_b2
#define SF_SHARED 640
#define WF_FEAT 0           // 16 rows x 16
#define WF_R    256
#define WF_AP   272
#define WF_CP   288
#define WF_AE   304
#define WF_CE   320
#define WF_DY   336
#define WF_DRP  352
#define WF_SUMP 368
#define WF_SQP  384
#define WF_SUME 400
#define WF_SQE  416
#define WF_DXP  432         // 16x16
#define WF_DXE  688         // 16x16
#define WF_SIZE 960
#define SF_TOTAL (SF_SHARED + WAVES * WF_SIZE)      // 4480 floats

// f16 section (half indices) -- weight slots hold SWIZZLED fragment images.
// Forward and transposed images are simultaneously resident.
#define SH_PW0  0            // 32x64    fwd image (2048 h)
#define SH_PW1  2048         // 64x64    fwd image (4096 h)
#define SH_EW0  6144         // 32x128   fwd image (4096 h)
#define SH_EW1  10240        // 128x128  fwd image (16384 h)
#define SH_PW0T 26624        // 64x16  W0^T image (1024 h)
#define SH_PW1T 27648        // 64x64  W1^T image (4096 h)
#define SH_EW0T 31744        // 128x16 W0^T image (2048 h)
#define SH_EW1T 33792        // 128x128 W1^T image (16384 h)
#define SH_SHARED 50176
#define WH_XP 0              // 16x32
#define WH_XE 512            // 16x32
#define WH_H0 1024           // 16x64
#define WH_H2 2048           // 16x64
#define WH_G0 3072           // 16x128
#define WH_G2 5120           // 16x128
#define WH_DA 7168           // 16x128
#define WH_DB 9216           // 16x128
#define WH_SIZE 11264
#define SH_TOTAL (SH_SHARED + WAVES * WH_SIZE)      // 95232 halves

#define SMEM_BYTES (SF_TOTAL * 4 + SH_TOTAL * 2)    // ~204 KB (<= 320 KB WGP LDS)

// -------------------- branchless fast math --------------------
__device__ __forceinline__ float exp_fast(float x) {          // e^x
  return __builtin_amdgcn_exp2f(x * LOG2E);
}
__device__ __forceinline__ float tanh_fast(float x) {
  float ax = __builtin_fabsf(x);
  float e  = exp_fast(ax + ax);
  float t  = 1.f - 2.f * __builtin_amdgcn_rcpf(e + 1.f);
  return __builtin_copysignf(t, x);
}

// ------------------------------------------------------------------
// Build a B-fragment-swizzled weight image in LDS.
// Logical B matrix is (K x N); element B[k][n] comes from
//   trans==false :  W[k*Ws + n]     (W is KxN row-major)
//   trans==true  :  W[n*Ws + k]     (transposed use)
// Zero outside realK/realN. Image layout: fragment f = nt*(K/32)+kt holds
// 32 lanes x 16 contiguous halves.
__device__ __forceinline__ void swizzle_fill(_Float16* dst, const float* W,
                                             int Ws, int K, int N,
                                             int realK, int realN, bool trans) {
  const int fpc = K >> 5;
  const int total = K * N;
  for (int idx = threadIdx.x; idx < total; idx += TPB) {
    int h    = idx & 15;
    int lane = (idx >> 4) & 31;
    int f    = idx >> 9;
    int kt   = f % fpc;
    int nt   = f / fpc;
    int k = kt * 32 + ((h & 8) << 1) + ((lane & 16) ? 8 : 0) + (h & 7);
    int n = nt * 16 + (lane & 15);
    float v = 0.f;
    if (k < realK && n < realN) v = trans ? W[n * Ws + k] : W[k * Ws + n];
    dst[idx] = (_Float16)v;
  }
}

// One 16x16 tile of D = A(16xK) @ B(KxN) using a swizzled weight image.
__device__ __forceinline__ v8f wmma16s(const _Float16* A, int As,
                                       const _Float16* Wsz, int K, int nt) {
  const int lane = threadIdx.x & 31;
  const int mrow = lane & 15;
  const int kA   = (lane & 16) ? 8 : 0;
  const int fpc  = K >> 5;
  v8f acc = {};
#pragma unroll
  for (int kt = 0; kt < 4; ++kt) {      // max K/32 == 4; guarded below
    if (kt >= fpc) break;
    const _Float16* Ap = A + mrow * As + kt * 32 + kA;
    v8h a0 = *(const v8h*)Ap;
    v8h a1 = *(const v8h*)(Ap + 16);
    v16h a = __builtin_shufflevector(a0, a1, 0, 1, 2, 3, 4, 5, 6, 7,
                                     8, 9, 10, 11, 12, 13, 14, 15);
    v16h b = *(const v16h*)(Wsz + ((nt * fpc + kt) * 32 + lane) * 16);
    acc = __builtin_amdgcn_wmma_f32_16x16x32_f16(false, a, false, b,
                                                 (short)0, acc, false, false);
  }
  return acc;
}

__device__ __forceinline__ void loadF(float* dst, const float* src, int n) {
  for (int i = threadIdx.x; i < n; i += TPB) dst[i] = src[i];
}

// -------------------- per-row feature builder --------------------
__device__ __forceinline__ void compute_features(int row,
    const float* __restrict__ dr, const float* __restrict__ ori,
    const float* __restrict__ nori,
    float* featw, float* Rout, _Float16* Xp, _Float16* Xe) {
  const int b = row / NB;
  const float* dp = dr + (size_t)row * 3;
  float d0 = dp[0] + EPSC, d1 = dp[1] + EPSC, d2 = dp[2] + EPSC;
  float R = sqrtf(d0 * d0 + d1 * d1 + d2 * d2);
  float inv = 1.f / R;
  d0 *= inv; d1 *= inv; d2 *= inv;
  const float* O  = ori  + (size_t)b * 9;
  const float* NO = nori + (size_t)row * 9;
  float f[15];
#pragma unroll
  for (int k = 0; k < 3; ++k) {
    f[k]     = d0 * O[k]  + d1 * O[3 + k]  + d2 * O[6 + k];
    f[3 + k] = d0 * NO[k] + d1 * NO[3 + k] + d2 * NO[6 + k];
  }
#pragma unroll
  for (int i = 0; i < 3; ++i)
#pragma unroll
    for (int j = 0; j < 3; ++j)
      f[6 + 3 * i + j] =
          O[i] * NO[j] + O[3 + i] * NO[3 + j] + O[6 + i] * NO[6 + j];
  *Rout = R;
  v8h z = {};
  *(v8h*)(Xp + 16) = z; *(v8h*)(Xp + 24) = z;
  *(v8h*)(Xe + 16) = z; *(v8h*)(Xe + 24) = z;
  Xp[15] = (_Float16)0.f;
  Xe[0]  = (_Float16)R;
#pragma unroll
  for (int t = 0; t < 15; ++t) {
    featw[t] = f[t];
    float s = f[t] * f[t];
    Xp[t] = (_Float16)s;
    Xe[1 + t] = (_Float16)s;
  }
}

// ==================== kernel 0: zero accumulators ====================
__global__ void init_zero_kernel(float* __restrict__ ws, float* __restrict__ out) {
  int i = blockIdx.x * blockDim.x + threadIdx.x;
  if (i < WS_TOTAL) ws[i] = 0.f;
  if (i < B_ * 3)   out[i] = 0.f;   // force region (atomics)
}

// ==================== kernel 1: BN statistics pass ====================
__global__ __launch_bounds__(TPB) void stats_kernel(
    const float* __restrict__ dr, const float* __restrict__ ori,
    const float* __restrict__ nori,
    const float* __restrict__ pW0, const float* __restrict__ pb0,
    const float* __restrict__ pW1, const float* __restrict__ pb1,
    const float* __restrict__ eW0, const float* __restrict__ eb0,
    const float* __restrict__ eW1, const float* __restrict__ eb1,
    float* __restrict__ ws) {
  extern __shared__ char smem_raw[];
  float*    sf = (float*)smem_raw;
  _Float16* sh = (_Float16*)(smem_raw + SF_TOTAL * 4);
  const int lane = threadIdx.x & 31, wave = threadIdx.x >> 5;
  float*    wf = sf + SF_SHARED + wave * WF_SIZE;
  _Float16* wh = sh + SH_SHARED + wave * WH_SIZE;

  __builtin_prefetch(eW1, 0, 1);
  swizzle_fill(sh + SH_PW0, pW0, 64, 32, 64, 15, 64, false);
  swizzle_fill(sh + SH_PW1, pW1, 64, 64, 64, 64, 64, false);
  swizzle_fill(sh + SH_EW0, eW0, 128, 32, 128, 16, 128, false);
  swizzle_fill(sh + SH_EW1, eW1, 128, 128, 128, 128, 128, false);
  loadF(sf + SF_PB0, pb0, 64);  loadF(sf + SF_PB1, pb1, 64);
  loadF(sf + SF_EB0, eb0, 128); loadF(sf + SF_EB1, eb1, 128);
  __syncthreads();

  const int mofs = (lane & 16) ? 8 : 0;
  const int tile0 = (blockIdx.x * WAVES + wave) * TPW * 16;

  for (int t = 0; t < TPW; ++t) {
    const int tile = tile0 + t * 16;
    if (lane < 16) {
      compute_features(tile + lane, dr, ori, nori, wf + WF_FEAT + lane * 16,
                       wf + WF_R + lane, wh + WH_XP + lane * 32,
                       wh + WH_XE + lane * 32);
      wf[WF_SUMP + lane] = 0.f; wf[WF_SQP + lane] = 0.f;
      wf[WF_SUME + lane] = 0.f; wf[WF_SQE + lane] = 0.f;
    }
    __syncthreads();
#pragma unroll
    for (int nt = 0; nt < 4; ++nt) {
      v8f acc = wmma16s(wh + WH_XP, 32, sh + SH_PW0, 32, nt);
      int n = nt * 16 + (lane & 15);
#pragma unroll
      for (int q = 0; q < 8; ++q)
        wh[WH_H0 + (mofs + q) * 64 + n] =
            (_Float16)tanh_fast(acc[q] + sf[SF_PB0 + n]);
    }
    __syncthreads();
#pragma unroll
    for (int nt = 0; nt < 4; ++nt) {
      v8f acc = wmma16s(wh + WH_H0, 64, sh + SH_PW1, 64, nt);
      int n = nt * 16 + (lane & 15);
#pragma unroll
      for (int q = 0; q < 8; ++q) {
        float v = acc[q] + sf[SF_PB1 + n];
        atomicAdd(&wf[WF_SUMP + mofs + q], v);
        atomicAdd(&wf[WF_SQP + mofs + q], v * v);
      }
    }
#pragma unroll
    for (int nt = 0; nt < 8; ++nt) {
      v8f acc = wmma16s(wh + WH_XE, 32, sh + SH_EW0, 32, nt);
      int n = nt * 16 + (lane & 15);
#pragma unroll
      for (int q = 0; q < 8; ++q)
        wh[WH_G0 + (mofs + q) * 128 + n] =
            (_Float16)tanh_fast(acc[q] + sf[SF_EB0 + n]);
    }
    __syncthreads();
#pragma unroll
    for (int nt = 0; nt < 8; ++nt) {
      v8f acc = wmma16s(wh + WH_G0, 128, sh + SH_EW1, 128, nt);
      int n = nt * 16 + (lane & 15);
#pragma unroll
      for (int q = 0; q < 8; ++q) {
        float v = acc[q] + sf[SF_EB1 + n];
        atomicAdd(&wf[WF_SUME + mofs + q], v);
        atomicAdd(&wf[WF_SQE + mofs + q], v * v);
      }
    }
    __syncthreads();
    if (lane < 16) {
      int ch = (tile + lane) % NB;
      atomicAdd(ws + WS_BNP_SUM + ch, wf[WF_SUMP + lane]);
      atomicAdd(ws + WS_BNP_SQ  + ch, wf[WF_SQP  + lane]);
      atomicAdd(ws + WS_BNE_SUM + ch, wf[WF_SUME + lane]);
      atomicAdd(ws + WS_BNE_SQ  + ch, wf[WF_SQE  + lane]);
    }
    __syncthreads();
  }
}

// ==================== kernel 2: BN finalize ====================
__global__ void bn_finalize_kernel(const float* __restrict__ pbng,
                                   const float* __restrict__ pbnb,
                                   const float* __restrict__ ebng,
                                   const float* __restrict__ ebnb,
                                   float* __restrict__ ws) {
  int ch = blockIdx.x * blockDim.x + threadIdx.x;
  if (ch >= NB) return;
  float cnt = (float)B_ * 64.f;
  float m = ws[WS_BNP_SUM + ch] / cnt;
  float v = ws[WS_BNP_SQ + ch] / cnt - m * m;
  float a = pbng[ch] * rsqrtf(v + BNEPS);
  ws[WS_BNP_A + ch] = a; ws[WS_BNP_C + ch] = pbnb[ch] - m * a;
  cnt = (float)B_ * 128.f;
  m = ws[WS_BNE_SUM + ch] / cnt;
  v = ws[WS_BNE_SQ + ch] / cnt - m * m;
  a = ebng[ch] * rsqrtf(v + BNEPS);
  ws[WS_BNE_A + ch] = a; ws[WS_BNE_C + ch] = ebnb[ch] - m * a;
}

// ==================== kernel 3: forward + analytic backward ====================
__global__ __launch_bounds__(TPB) void fwdbwd_kernel(
    const float* __restrict__ dr, const float* __restrict__ ori,
    const float* __restrict__ nori,
    const float* __restrict__ pW0, const float* __restrict__ pb0,
    const float* __restrict__ pW1, const float* __restrict__ pb1,
    const float* __restrict__ pW2, const float* __restrict__ pb2,
    const float* __restrict__ eW0, const float* __restrict__ eb0,
    const float* __restrict__ eW1, const float* __restrict__ eb1,
    const float* __restrict__ eW2, const float* __restrict__ eb2,
    const float* __restrict__ f1p, const float* __restrict__ f2p,
    float* __restrict__ ws, float* __restrict__ out) {
  extern __shared__ char smem_raw[];
  float*    sf = (float*)smem_raw;
  _Float16* sh = (_Float16*)(smem_raw + SF_TOTAL * 4);
  const int lane = threadIdx.x & 31, wave = threadIdx.x >> 5;
  float*    wf = sf + SF_SHARED + wave * WF_SIZE;
  _Float16* wh = sh + SH_SHARED + wave * WH_SIZE;

  __builtin_prefetch(eW1, 0, 1);
  // forward images
  swizzle_fill(sh + SH_PW0, pW0, 64, 32, 64, 15, 64, false);
  swizzle_fill(sh + SH_PW1, pW1, 64, 64, 64, 64, 64, false);
  swizzle_fill(sh + SH_EW0, eW0, 128, 32, 128, 16, 128, false);
  swizzle_fill(sh + SH_EW1, eW1, 128, 128, 128, 128, 128, false);
  // transposed images (resident simultaneously; no mid-kernel refill)
  swizzle_fill(sh + SH_PW0T, pW0, 64, 64, 16, 64, 15, true);
  swizzle_fill(sh + SH_PW1T, pW1, 64, 64, 64, 64, 64, true);
  swizzle_fill(sh + SH_EW0T, eW0, 128, 128, 16, 128, 16, true);
  swizzle_fill(sh + SH_EW1T, eW1, 128, 128, 128, 128, 128, true);
  loadF(sf + SF_PB0, pb0, 64);  loadF(sf + SF_PB1, pb1, 64);
  loadF(sf + SF_EB0, eb0, 128); loadF(sf + SF_EB1, eb1, 128);
  loadF(sf + SF_PW2, pW2, 64);  loadF(sf + SF_EW2, eW2, 128);
  if (threadIdx.x == 0) { sf[SF_SC] = pb2[0]; sf[SF_SC + 1] = eb2[0]; }
  __syncthreads();

  const int mofs = (lane & 16) ? 8 : 0;
  const int tile0 = (blockIdx.x * WAVES + wave) * TPW * 16;

  for (int t = 0; t < TPW; ++t) {
    const int tile = tile0 + t * 16;
    if (lane < 16) {
      int row = tile + lane, ch = row % NB;
      compute_features(row, dr, ori, nori, wf + WF_FEAT + lane * 16,
                       wf + WF_R + lane, wh + WH_XP + lane * 32,
                       wh + WH_XE + lane * 32);
      wf[WF_AP + lane] = ws[WS_BNP_A + ch];
      wf[WF_CP + lane] = ws[WS_BNP_C + ch];
      wf[WF_AE + lane] = ws[WS_BNE_A + ch];
      wf[WF_CE + lane] = ws[WS_BNE_C + ch];
    }
    __syncthreads();
    // ---- forward p MLP ----
#pragma unroll
    for (int nt = 0; nt < 4; ++nt) {
      v8f acc = wmma16s(wh + WH_XP, 32, sh + SH_PW0, 32, nt);
      int n = nt * 16 + (lane & 15);
#pragma unroll
      for (int q = 0; q < 8; ++q)
        wh[WH_H0 + (mofs + q) * 64 + n] =
            (_Float16)tanh_fast(acc[q] + sf[SF_PB0 + n]);
    }
    __syncthreads();
#pragma unroll
    for (int nt = 0; nt < 4; ++nt) {
      v8f acc = wmma16s(wh + WH_H0, 64, sh + SH_PW1, 64, nt);
      int n = nt * 16 + (lane & 15);
#pragma unroll
      for (int q = 0; q < 8; ++q) {
        int m = mofs + q;
        float v = acc[q] + sf[SF_PB1 + n];
        wh[WH_H2 + m * 64 + n] =
            (_Float16)tanh_fast(wf[WF_AP + m] * v + wf[WF_CP + m]);
      }
    }
    // ---- forward e MLP ----
#pragma unroll
    for (int nt = 0; nt < 8; ++nt) {
      v8f acc = wmma16s(wh + WH_XE, 32, sh + SH_EW0, 32, nt);
      int n = nt * 16 + (lane & 15);
#pragma unroll
      for (int q = 0; q < 8; ++q)
        wh[WH_G0 + (mofs + q) * 128 + n] =
            (_Float16)tanh_fast(acc[q] + sf[SF_EB0 + n]);
    }
    __syncthreads();
#pragma unroll
    for (int nt = 0; nt < 8; ++nt) {
      v8f acc = wmma16s(wh + WH_G0, 128, sh + SH_EW1, 128, nt);
      int n = nt * 16 + (lane & 15);
#pragma unroll
      for (int q = 0; q < 8; ++q) {
        int m = mofs + q;
        float v = acc[q] + sf[SF_EB1 + n];
        wh[WH_G2 + m * 128 + n] =
            (_Float16)tanh_fast(wf[WF_AE + m] * v + wf[WF_CE + m]);
      }
    }
    __syncthreads();
    // ---- per-row heads, prior, and head gradients ----
    if (lane < 16) {
      int r = lane, row = tile + r, b = row / NB;
      float y = sf[SF_SC];
#pragma unroll
      for (int j = 0; j < 64; j += 8) {
        v8h hv = *(const v8h*)(wh + WH_H2 + r * 64 + j);
#pragma unroll
        for (int q = 0; q < 8; ++q) y += (float)hv[q] * sf[SF_PW2 + j + q];
      }
      float sym = sf[SF_SC + 1];
#pragma unroll
      for (int j = 0; j < 128; j += 8) {
        v8h gv = *(const v8h*)(wh + WH_G2 + r * 128 + j);
#pragma unroll
        for (int q = 0; q < 8; ++q) sym += (float)gv[q] * sf[SF_EW2 + j + q];
      }
      float enc0 = tanh_fast(y);
      float EC = enc0 * enc0 + EPSC;
      float base = wf[WF_R + r] - EC;
      float ps = 0.f, db = 0.f;
#pragma unroll
      for (int k = 0; k < 3; ++k) {
        float f1s = f1p[k] * f1p[k] + EPSC;
        float f2s = f2p[k] * f2p[k] + EPSC;
        float tt = f1s * base;
        // p = tt^(-f2s) = exp2(-f2s * log2(tt))
        float p = __builtin_amdgcn_exp2f(-f2s * __builtin_amdgcn_logf(tt));
        ps += p;
        db -= f2s * f1s * p * __builtin_amdgcn_rcpf(tt);  // d/dbase
      }
      float dEdb = 200.f * db;
      wf[WF_DRP + r] = dEdb;                                   // dE/dR (prior)
      wf[WF_DY + r] = (-dEdb) * 2.f * enc0 * (1.f - enc0 * enc0);  // dE/dy
      atomicAdd(ws + WS_PRIOR + b, ps);
      atomicAdd(ws + WS_S, sym);
    }
    __syncthreads();
    // ---- backward p: dh1 -> dh0 -> dEcin_p ----
    {
      int m = lane >> 1, jb = (lane & 1) * 32;
      float s = wf[WF_DY + m] * wf[WF_AP + m];
#pragma unroll
      for (int c = 0; c < 32; c += 8) {
        int j = jb + c;
        v8h h2v = *(const v8h*)(wh + WH_H2 + m * 64 + j);
        v8h o;
#pragma unroll
        for (int q = 0; q < 8; ++q) {
          float h2 = (float)h2v[q];
          o[q] = (_Float16)(s * sf[SF_PW2 + j + q] * (1.f - h2 * h2));
        }
        *(v8h*)(wh + WH_DA + m * 64 + j) = o;
      }
    }
    __syncthreads();
#pragma unroll
    for (int nt = 0; nt < 4; ++nt) {
      v8f acc = wmma16s(wh + WH_DA, 64, sh + SH_PW1T, 64, nt);
      int n = nt * 16 + (lane & 15);
#pragma unroll
      for (int q = 0; q < 8; ++q) {
        int m = mofs + q;
        float h0 = (float)wh[WH_H0 + m * 64 + n];
        wh[WH_DB + m * 64 + n] = (_Float16)(acc[q] * (1.f - h0 * h0));
      }
    }
    __syncthreads();
    {
      v8f acc = wmma16s(wh + WH_DB, 64, sh + SH_PW0T, 64, 0);
      int n = lane & 15;
#pragma unroll
      for (int q = 0; q < 8; ++q) wf[WF_DXP + (mofs + q) * 16 + n] = acc[q];
    }
    __syncthreads();
    // ---- backward e: dg1 -> dg0 -> dxe  (dE/dsym = B) ----
    {
      int m = lane >> 1, jb = (lane & 1) * 64;
      float s = (float)B_ * wf[WF_AE + m];
#pragma unroll
      for (int c = 0; c < 64; c += 8) {
        int j = jb + c;
        v8h g2v = *(const v8h*)(wh + WH_G2 + m * 128 + j);
        v8h o;
#pragma unroll
        for (int q = 0; q < 8; ++q) {
          float g2 = (float)g2v[q];
          o[q] = (_Float16)(s * sf[SF_EW2 + j + q] * (1.f - g2 * g2));
        }
        *(v8h*)(wh + WH_DA + m * 128 + j) = o;
      }
    }
    __syncthreads();
#pragma unroll
    for (int nt = 0; nt < 8; ++nt) {
      v8f acc = wmma16s(wh + WH_DA, 128, sh + SH_EW1T, 128, nt);
      int n = nt * 16 + (lane & 15);
#pragma unroll
      for (int q = 0; q < 8; ++q) {
        int m = mofs + q;
        float g0 = (float)wh[WH_G0 + m * 128 + n];
        wh[WH_DB + m * 128 + n] = (_Float16)(acc[q] * (1.f - g0 * g0));
      }
    }
    __syncthreads();
    {
      v8f acc = wmma16s(wh + WH_DB, 128, sh + SH_EW0T, 128, 0);
      int n = lane & 15;
#pragma unroll
      for (int q = 0; q < 8; ++q) wf[WF_DXE + (mofs + q) * 16 + n] = acc[q];
    }
    __syncthreads();
    // ---- per-row chain to dr / orientation ----
    if (lane < 16) {
      int r = lane, row = tile + r, b = row / NB;
      const float* dp = dr + (size_t)row * 3;
      float d0 = dp[0] + EPSC, d1 = dp[1] + EPSC, d2 = dp[2] + EPSC;
      float R = wf[WF_R + r];
      float inv = 1.f / R;
      d0 *= inv; d1 *= inv; d2 *= inv;
      const float* O  = ori  + (size_t)b * 9;
      const float* NO = nori + (size_t)row * 9;
      float df[15];
#pragma unroll
      for (int u = 0; u < 15; ++u) {
        float dE = wf[WF_DXP + r * 16 + u] + wf[WF_DXE + r * 16 + 1 + u];
        df[u] = 2.f * wf[WF_FEAT + r * 16 + u] * dE;   // d(feat^2)
      }
      float dRt = wf[WF_DXE + r * 16 + 0] + wf[WF_DRP + r];
      float dvec[3] = {d0, d1, d2};
      float dd[3];
#pragma unroll
      for (int c = 0; c < 3; ++c)
        dd[c] = O[c * 3 + 0] * df[0] + O[c * 3 + 1] * df[1] + O[c * 3 + 2] * df[2] +
                NO[c * 3 + 0] * df[3] + NO[c * 3 + 1] * df[4] + NO[c * 3 + 2] * df[5];
      float ddot = dd[0] * d0 + dd[1] * d1 + dd[2] * d2;
#pragma unroll
      for (int c = 0; c < 3; ++c) {
        float g = (dd[c] - ddot * dvec[c]) * inv + dRt * dvec[c];
        atomicAdd(out + (size_t)b * 3 + c, g);         // force accumulation
      }
#pragma unroll
      for (int c = 0; c < 3; ++c)
#pragma unroll
        for (int k = 0; k < 3; ++k) {
          float go = dvec[c] * df[k];
#pragma unroll
          for (int j = 0; j < 3; ++j) go += NO[c * 3 + j] * df[6 + 3 * k + j];
          atomicAdd(ws + WS_GO + (size_t)b * 9 + c * 3 + k, go);
        }
    }
    __syncthreads();
  }
}

// ==================== kernel 4: pe + torque ====================
__global__ void finalize_kernel(const float* __restrict__ ori,
                                const float* __restrict__ ws,
                                float* __restrict__ out) {
  int b = blockIdx.x * blockDim.x + threadIdx.x;
  if (b >= B_) return;
  out[2 * B_ * 3 + b] = ws[WS_S] + 200.f * ws[WS_PRIOR + b];  // pe
  const float* gO = ws + WS_GO + (size_t)b * 9;
  const float* O  = ori + (size_t)b * 9;
  float t0 = 0.f, t1 = 0.f, t2 = 0.f;
#pragma unroll
  for (int k = 0; k < 3; ++k) {
    float u0 = gO[0 + k], u1 = gO[3 + k], u2 = gO[6 + k];
    float v0 = O[0 + k],  v1 = O[3 + k],  v2 = O[6 + k];
    t0 += u1 * v2 - u2 * v1;
    t1 += u2 * v0 - u0 * v2;
    t2 += u0 * v1 - u1 * v0;
  }
  out[B_ * 3 + b * 3 + 0] = t0;
  out[B_ * 3 + b * 3 + 1] = t1;
  out[B_ * 3 + b * 3 + 2] = t2;
}

// ==================== host launcher ====================
extern "C" void kernel_launch(void* const* d_in, const int* in_sizes, int n_in,
                              void* d_out, int out_size, void* d_ws, size_t ws_size,
                              hipStream_t stream) {
  (void)in_sizes; (void)n_in; (void)out_size; (void)ws_size;
  const float* dr   = (const float*)d_in[0];
  const float* ori  = (const float*)d_in[1];
  const float* nori = (const float*)d_in[2];
  const float* pW0  = (const float*)d_in[3];
  const float* pb0  = (const float*)d_in[4];
  const float* pW1  = (const float*)d_in[5];
  const float* pb1  = (const float*)d_in[6];
  const float* pbng = (const float*)d_in[7];
  const float* pbnb = (const float*)d_in[8];
  const float* pW2  = (const float*)d_in[9];
  const float* pb2  = (const float*)d_in[10];
  const float* eW0  = (const float*)d_in[11];
  const float* eb0  = (const float*)d_in[12];
  const float* eW1  = (const float*)d_in[13];
  const float* eb1  = (const float*)d_in[14];
  const float* ebng = (const float*)d_in[15];
  const float* ebnb = (const float*)d_in[16];
  const float* eW2  = (const float*)d_in[17];
  const float* eb2  = (const float*)d_in[18];
  const float* f1   = (const float*)d_in[19];
  const float* f2   = (const float*)d_in[20];
  float* out = (float*)d_out;
  float* ws  = (float*)d_ws;

  (void)hipFuncSetAttribute(reinterpret_cast<const void*>(stats_kernel),
                            hipFuncAttributeMaxDynamicSharedMemorySize, SMEM_BYTES);
  (void)hipFuncSetAttribute(reinterpret_cast<const void*>(fwdbwd_kernel),
                            hipFuncAttributeMaxDynamicSharedMemorySize, SMEM_BYTES);

  init_zero_kernel<<<(WS_TOTAL + 255) / 256, 256, 0, stream>>>(ws, out);
  stats_kernel<<<NBLOCKS, TPB, SMEM_BYTES, stream>>>(
      dr, ori, nori, pW0, pb0, pW1, pb1, eW0, eb0, eW1, eb1, ws);
  bn_finalize_kernel<<<1, 256, 0, stream>>>(pbng, pbnb, ebng, ebnb, ws);
  fwdbwd_kernel<<<NBLOCKS, TPB, SMEM_BYTES, stream>>>(
      dr, ori, nori, pW0, pb0, pW1, pb1, pW2, pb2,
      eW0, eb0, eW1, eb1, eW2, eb2, f1, f2, ws, out);
  finalize_kernel<<<(B_ + 255) / 256, 256, 0, stream>>>(ori, ws, out);
}
